// SelfAttention_47012712022545
// MI455X (gfx1250) — compile-verified, matrix operand order checked by reading, not run on hardware
//
#include <hip/hip_runtime.h>
#include <hip/hip_bf16.h>
#include <stdint.h>

#define B_  8
#define C_  512
#define N_  1024
#define H_  8
#define D_  64
#define PE_ 3969   // (2*32-1)^2

typedef __attribute__((ext_vector_type(16))) __bf16 v16bf;
typedef __attribute__((ext_vector_type(8)))  __bf16 v8bf;
typedef __attribute__((ext_vector_type(8)))  float  v8f;

static __device__ __forceinline__ __bf16 f2bf(float f) {
  uint32_t u = __builtin_bit_cast(uint32_t, f);
  u += 0x7FFFu + ((u >> 16) & 1u);
  unsigned short h = (unsigned short)(u >> 16);
  return __builtin_bit_cast(__bf16, h);
}

// 16-bit A/B fragment per CDNA5 ISA 7.12.2: lane holds 16 bf16 as two
// contiguous 8-element (16B) chunks at +0 and +16 elements.
static __device__ __forceinline__ v16bf ld_frag(const __bf16* p) {
  v16bf r;
  *(v8bf*)&r        = *(const v8bf*)(p);
  *(((v8bf*)&r) + 1) = *(const v8bf*)(p + 16);
  return r;
}

static __device__ __forceinline__ v8f wmma_bf16(v16bf a, v16bf b, v8f c) {
  return __builtin_amdgcn_wmma_f32_16x16x32_bf16(false, a, false, b, (short)0, c,
                                                 false, false);
}

// ---------------------------------------------------------------- K0: weights
__global__ void k_cvt_weights(const float* __restrict__ wq, const float* __restrict__ wk,
                              const float* __restrict__ wv, const float* __restrict__ wp,
                              __bf16* __restrict__ o) {
  int idx = blockIdx.x * blockDim.x + threadIdx.x;   // 4 * 262144 total
  int mat = idx >> 18;
  int off = idx & 0x3FFFF;
  const float* s = (mat == 0) ? wq : (mat == 1) ? wk : (mat == 2) ? wv : wp;
  o[idx] = f2bf(s[off]);
}

// -------------------------------------------------------------- K1: groupnorm
// One block per (group, batch). A group (16 channels x 1024 px) is contiguous.
__global__ void k_groupnorm(const float* __restrict__ x, const float* __restrict__ gamma,
                            const float* __restrict__ beta, __bf16* __restrict__ xnT) {
  int g = blockIdx.x, b = blockIdx.y;
  const float* xb = x + ((size_t)b * C_ + (size_t)g * 16) * N_;
  float s = 0.f, ss = 0.f;
  for (int i = threadIdx.x; i < 16 * N_; i += 256) {
    float v = xb[i];
    s += v; ss += v * v;
  }
  __shared__ float rs[256], rss[256];
  rs[threadIdx.x] = s; rss[threadIdx.x] = ss;
  __syncthreads();
  for (int st = 128; st > 0; st >>= 1) {
    if ((int)threadIdx.x < st) {
      rs[threadIdx.x]  += rs[threadIdx.x + st];
      rss[threadIdx.x] += rss[threadIdx.x + st];
    }
    __syncthreads();
  }
  float mean = rs[0] * (1.0f / 16384.f);
  float var  = rss[0] * (1.0f / 16384.f) - mean * mean;
  float rinv = rsqrtf(var + 1e-6f);
  for (int i = threadIdx.x; i < 16 * N_; i += 256) {
    int c = g * 16 + (i >> 10);
    int n = i & 1023;
    float v = (xb[i] - mean) * rinv * gamma[c] + beta[c];
    xnT[((size_t)b * N_ + n) * C_ + c] = f2bf(v);   // transposed [b][n][c]
  }
}

// -------------------------------------------------------------- K2: QKV GEMM
// One wave: 16(o) x 64(n) tile. grid = (N/64, C/128, B*3), block = 256 (8 waves)
__global__ void k_qkv(const __bf16* __restrict__ wqB, const __bf16* __restrict__ wkB,
                      const __bf16* __restrict__ wvB,
                      const float* __restrict__ bq, const float* __restrict__ bk,
                      const float* __restrict__ bv,
                      const __bf16* __restrict__ xnT,
                      __bf16* __restrict__ qb, __bf16* __restrict__ kb,
                      __bf16* __restrict__ vT) {
  int lane = threadIdx.x & 31, wave = threadIdx.x >> 5;
  int half = lane >> 4, lm = lane & 15;
  int mat = blockIdx.z % 3;
  int b   = blockIdx.z / 3;
  int m0  = ((int)blockIdx.y * 8 + wave) * 16;
  int n0  = (int)blockIdx.x * 64;
  const __bf16* W    = (mat == 0) ? wqB : (mat == 1) ? wkB : wvB;
  const float*  bias = (mat == 0) ? bq  : (mat == 1) ? bk  : bv;

  float bvals[8];
#pragma unroll
  for (int g = 0; g < 8; ++g) bvals[g] = bias[m0 + g + 8 * half];

  v8f a0 = {}, a1 = {}, a2 = {}, a3 = {};
  const __bf16* xb    = xnT + (size_t)b * N_ * C_;
  const __bf16* wrow  = W + (size_t)(m0 + lm) * C_ + half * 8;
  const __bf16* brow  = xb + (size_t)(n0 + lm) * C_ + half * 8;
  for (int k = 0; k < C_; k += 32) {
    __builtin_prefetch(brow + k + 32, 0, 1);
    v16bf a = ld_frag(wrow + k);
    a0 = wmma_bf16(a, ld_frag(brow + k), a0);
    a1 = wmma_bf16(a, ld_frag(brow + k + 16 * C_), a1);
    a2 = wmma_bf16(a, ld_frag(brow + k + 32 * C_), a2);
    a3 = wmma_bf16(a, ld_frag(brow + k + 48 * C_), a3);
  }

  v8f accs[4] = {a0, a1, a2, a3};
  int h = m0 >> 6;
  int dbase = (m0 & 63) + 8 * half;
  float scale = (mat == 0) ? 0.125f : 1.0f;    // 1/sqrt(64), folded into Q
  if (mat < 2) {
    __bf16* outp = (mat == 0) ? qb : kb;       // [b][h][n][d]
#pragma unroll
    for (int j = 0; j < 4; ++j) {
      int n = n0 + j * 16 + lm;
      v8bf pk;
#pragma unroll
      for (int g = 0; g < 8; ++g) pk[g] = f2bf((accs[j][g] + bvals[g]) * scale);
      *(v8bf*)(outp + ((((size_t)b * H_ + h) * N_ + n) * D_ + dbase)) = pk;
    }
  } else {                                     // V transposed: [b][h][d][n]
#pragma unroll
    for (int j = 0; j < 4; ++j) {
      int n = n0 + j * 16 + lm;
#pragma unroll
      for (int g = 0; g < 8; ++g) {
        int d = dbase + g;
        vT[(((size_t)b * H_ + h) * D_ + d) * N_ + n] = f2bf(accs[j][g] + bvals[g]);
      }
    }
  }
}

// --------------------------------------------------------- K3: flash attention
// One wave owns 16 query rows of one (b,h); streams 32 keys/iter.
__global__ void k_flash(const __bf16* __restrict__ Q, const __bf16* __restrict__ Kb,
                        const __bf16* __restrict__ Vt, const float* __restrict__ pos,
                        __bf16* __restrict__ aT) {
  __shared__ __align__(16) __bf16 Plds[4][16 * 32];
  int lane = threadIdx.x & 31, wave = threadIdx.x >> 5;
  int half = lane >> 4, lm = lane & 15;
  int gidx = (int)blockIdx.x * 4 + wave;      // 0..4095
  int qt = gidx & 63;
  int bh = gidx >> 6;
  int h = bh & 7, b = bh >> 3;
  int q0 = qt * 16;
  const __bf16* Qb = Q  + ((size_t)bh * N_) * D_;
  const __bf16* Kx = Kb + ((size_t)bh * N_) * D_;
  const __bf16* Vb = Vt + ((size_t)bh * D_) * N_;
  const float*  pe = pos + (size_t)h * PE_;

  // Q A-fragments (d = 0..31 and 32..63), already scaled by 1/8
  v16bf qa0 = ld_frag(Qb + (size_t)(q0 + lm) * D_ + half * 8);
  v16bf qa1 = ld_frag(Qb + (size_t)(q0 + lm) * D_ + 32 + half * 8);

  float run_max[8], run_sum[8];
  int qy[8], qx[8];
#pragma unroll
  for (int g = 0; g < 8; ++g) {
    run_max[g] = -1e30f; run_sum[g] = 0.f;
    int q = q0 + g + 8 * half;
    qy[g] = q >> 5; qx[g] = q & 31;
  }
  v8f o0 = {}, o1 = {}, o2 = {}, o3 = {};

  for (int kt = 0; kt < N_; kt += 32) {
    // ---- S = Q K^T (two 16-key tiles) ----
    const __bf16* kb0 = Kx + (size_t)(kt + lm) * D_ + half * 8;
    v8f s0 = {}, s1 = {};
    s0 = wmma_bf16(qa0, ld_frag(kb0), s0);
    s0 = wmma_bf16(qa1, ld_frag(kb0 + 32), s0);
    s1 = wmma_bf16(qa0, ld_frag(kb0 + 16 * D_), s1);
    s1 = wmma_bf16(qa1, ld_frag(kb0 + 16 * D_ + 32), s1);

    // ---- relative position bias ----
    int k0i = kt + lm,  ky0 = k0i >> 5, kx0 = k0i & 31;
    int k1i = k0i + 16, ky1 = k1i >> 5, kx1 = k1i & 31;
#pragma unroll
    for (int g = 0; g < 8; ++g) {
      s0[g] += pe[(qy[g] - ky0 + 31) * 63 + (qx[g] - kx0 + 31)];
      s1[g] += pe[(qy[g] - ky1 + 31) * 63 + (qx[g] - kx1 + 31)];
    }

    // ---- online softmax: rows live across 16-lane half groups ----
    float nm[8], alpha[8];
#pragma unroll
    for (int g = 0; g < 8; ++g) {
      float m = fmaxf(s0[g], s1[g]);
#pragma unroll
      for (int off = 8; off > 0; off >>= 1) m = fmaxf(m, __shfl_xor(m, off, 16));
      nm[g] = fmaxf(run_max[g], m);
      alpha[g] = __expf(run_max[g] - nm[g]);
      run_max[g] = nm[g];
    }
#pragma unroll
    for (int g = 0; g < 8; ++g) {
      s0[g] = __expf(s0[g] - nm[g]);
      s1[g] = __expf(s1[g] - nm[g]);
      float rsum = s0[g] + s1[g];
#pragma unroll
      for (int off = 8; off > 0; off >>= 1) rsum += __shfl_xor(rsum, off, 16);
      run_sum[g] = run_sum[g] * alpha[g] + rsum;
      o0[g] *= alpha[g]; o1[g] *= alpha[g]; o2[g] *= alpha[g]; o3[g] *= alpha[g];
    }

    // ---- P (f32, C/D layout) -> LDS -> bf16 A-layout fragment ----
    __bf16* pl = Plds[wave];
#pragma unroll
    for (int g = 0; g < 8; ++g) {
      int m = g + 8 * half;
      pl[m * 32 + lm]      = f2bf(s0[g]);
      pl[m * 32 + 16 + lm] = f2bf(s1[g]);
    }
    asm volatile("s_wait_dscnt 0" ::: "memory");
    v16bf pA = ld_frag(pl + lm * 32 + half * 8);

    // ---- O += P * V  (V transposed: [d][n], contiguous along keys) ----
    const __bf16* vb = Vb + (size_t)lm * N_ + kt + half * 8;
    o0 = wmma_bf16(pA, ld_frag(vb), o0);
    o1 = wmma_bf16(pA, ld_frag(vb + 16 * N_), o1);
    o2 = wmma_bf16(pA, ld_frag(vb + 32 * N_), o2);
    o3 = wmma_bf16(pA, ld_frag(vb + 48 * N_), o3);
  }

  // ---- finalize: divide by row sums, store transposed [b][n][c] bf16 ----
  float inv[8];
#pragma unroll
  for (int g = 0; g < 8; ++g) inv[g] = 1.0f / run_sum[g];
  v8f outs[4] = {o0, o1, o2, o3};
#pragma unroll
  for (int j = 0; j < 4; ++j) {
    int c = h * D_ + j * 16 + lm;
#pragma unroll
    for (int g = 0; g < 8; ++g) {
      int q = q0 + g + 8 * half;
      aT[((size_t)b * N_ + q) * C_ + c] = f2bf(outs[j][g] * inv[g]);
    }
  }
}

// ------------------------------------------------- K4: projection + residual
__global__ void k_proj(const __bf16* __restrict__ wpB, const float* __restrict__ bp,
                       const __bf16* __restrict__ aT, const float* __restrict__ x,
                       float* __restrict__ out) {
  int lane = threadIdx.x & 31, wave = threadIdx.x >> 5;
  int half = lane >> 4, lm = lane & 15;
  int b  = blockIdx.z;
  int m0 = ((int)blockIdx.y * 8 + wave) * 16;
  int n0 = (int)blockIdx.x * 64;

  float bvals[8];
#pragma unroll
  for (int g = 0; g < 8; ++g) bvals[g] = bp[m0 + g + 8 * half];

  v8f a0 = {}, a1 = {}, a2 = {}, a3 = {};
  const __bf16* ab   = aT + (size_t)b * N_ * C_;
  const __bf16* wrow = wpB + (size_t)(m0 + lm) * C_ + half * 8;
  const __bf16* brow = ab + (size_t)(n0 + lm) * C_ + half * 8;
  for (int k = 0; k < C_; k += 32) {
    __builtin_prefetch(brow + k + 32, 0, 1);
    v16bf a = ld_frag(wrow + k);
    a0 = wmma_bf16(a, ld_frag(brow + k), a0);
    a1 = wmma_bf16(a, ld_frag(brow + k + 16 * C_), a1);
    a2 = wmma_bf16(a, ld_frag(brow + k + 32 * C_), a2);
    a3 = wmma_bf16(a, ld_frag(brow + k + 48 * C_), a3);
  }
  v8f accs[4] = {a0, a1, a2, a3};
#pragma unroll
  for (int j = 0; j < 4; ++j) {
    int n = n0 + j * 16 + lm;
#pragma unroll
    for (int g = 0; g < 8; ++g) {
      int o = m0 + g + 8 * half;
      size_t oi = ((size_t)b * C_ + o) * N_ + n;
      out[oi] = accs[j][g] + bvals[g] + x[oi];
    }
  }
}

// ----------------------------------------------------------------------------
extern "C" void kernel_launch(void* const* d_in, const int* in_sizes, int n_in,
                              void* d_out, int out_size, void* d_ws, size_t ws_size,
                              hipStream_t stream) {
  const float* x     = (const float*)d_in[0];
  const float* gamma = (const float*)d_in[3];
  const float* beta  = (const float*)d_in[4];
  const float* wq    = (const float*)d_in[5];
  const float* bq    = (const float*)d_in[6];
  const float* wk    = (const float*)d_in[7];
  const float* bk    = (const float*)d_in[8];
  const float* wv    = (const float*)d_in[9];
  const float* bv    = (const float*)d_in[10];
  const float* wproj = (const float*)d_in[11];
  const float* bproj = (const float*)d_in[12];
  const float* pos   = (const float*)d_in[13];

  // workspace layout (bf16 elements)
  __bf16* wqB = (__bf16*)d_ws;            // 4 x 262144 weights
  __bf16* wkB = wqB + 262144;
  __bf16* wvB = wkB + 262144;
  __bf16* wpB = wvB + 262144;
  __bf16* xnT = wpB + 262144;             // [B][N][C]   8 MB
  __bf16* aT  = xnT + (size_t)B_ * N_ * C_;  // [B][N][C]
  __bf16* qb  = aT  + (size_t)B_ * N_ * C_;  // [B][H][N][D]
  __bf16* kb  = qb  + (size_t)B_ * N_ * C_;  // [B][H][N][D]
  __bf16* vT  = kb  + (size_t)B_ * N_ * C_;  // [B][H][D][N]

  k_cvt_weights<<<dim3(4096), 256, 0, stream>>>(wq, wk, wv, wproj, wqB);
  k_groupnorm<<<dim3(32, B_), 256, 0, stream>>>(x, gamma, beta, xnT);
  k_qkv<<<dim3(N_ / 64, C_ / 128, B_ * 3), 256, 0, stream>>>(
      wqB, wkB, wvB, bq, bk, bv, xnT, qb, kb, vT);
  k_flash<<<dim3(B_ * H_ * (N_ / 16) / 4), 128, 0, stream>>>(qb, kb, vT, pos, aT);
  k_proj<<<dim3(N_ / 64, C_ / 128, B_), 256, 0, stream>>>(
      wpB, bproj, aT, x, (float*)d_out);
}
